// PhasedClassifier_76768245448711
// MI455X (gfx1250) — matrix operand order repaired and checked by reading, not compiled, not run on hardware
//
#include <hip/hip_runtime.h>
#include <cstddef>

typedef __attribute__((ext_vector_type(16))) _Float16 v16h;
typedef __attribute__((ext_vector_type(8)))  float    v8f;

#define UNITS   128
#define NCLS    10
#define TSTEPS  1000
#define BATCH   256
#define ZW      512            // 4*UNITS
#define R_ON    0.05f
#define ALPHA_L 0.001f
#define LN_EPS  1e-3f

#define BT      16             // batch rows per workgroup
#define NWG     (BATCH/BT)     // 16 workgroups
#define NT0     32             // 512/16 N-tiles
#define KT0     4              // 128/32 K-tiles (layer0 / FC)
#define KT1     8              // 256/32 K-tiles (layer1 stacked)

#define WH0F_ELEMS (NT0*KT0*32*16)   // 65536  f16
#define W1F_ELEMS  (NT0*KT1*32*16)   // 131072 f16
#define WFCF_ELEMS (KT0*32*16)       // 2048   f16

__device__ __forceinline__ float sigf(float x)   { return 1.f/(1.f+__expf(-x)); }
__device__ __forceinline__ float tanhft(float x) { return 2.f/(1.f+__expf(-2.f*x)) - 1.f; }

// -------- Prep: swizzle fp32 weights into per-lane WMMA-B fragment order (f16) -------
// B fragment (32x16, f16), k order used here: lane = g*16 + n (g=lane>>4), half j holds B[g*16+j][n].
// Layout in memory: [nt][kt][lane][16 halves] -> lane's 16 halves contiguous (32B).
__global__ void plstm_prep(const float* __restrict__ Wh0, const float* __restrict__ Wx1,
                           const float* __restrict__ Wh1, const float* __restrict__ Wfc,
                           _Float16* __restrict__ wh0f, _Float16* __restrict__ w1f,
                           _Float16* __restrict__ wfcf) {
  const int i = blockIdx.x * blockDim.x + threadIdx.x;
  if (i < WH0F_ELEMS) {
    const int j = i & 15, lane = (i >> 4) & 31, ktnt = i >> 9;
    const int kt = ktnt & 3, nt = ktnt >> 2;
    const int k   = kt*32 + (lane >> 4)*16 + j;
    const int col = nt*16 + (lane & 15);
    wh0f[i] = (_Float16)Wh0[k*ZW + col];
  }
  const int i1 = i - WH0F_ELEMS;
  if (i1 >= 0 && i1 < W1F_ELEMS) {
    const int j = i1 & 15, lane = (i1 >> 4) & 31, ktnt = i1 >> 9;
    const int kt = ktnt & 7, nt = ktnt >> 3;
    const int k   = kt*32 + (lane >> 4)*16 + j;     // 0..255 stacked [Wx1; Wh1]
    const int col = nt*16 + (lane & 15);
    const float v = (k < UNITS) ? Wx1[k*ZW + col] : Wh1[(k-UNITS)*ZW + col];
    w1f[i1] = (_Float16)v;
  }
  const int i2 = i - WH0F_ELEMS - W1F_ELEMS;
  if (i2 >= 0 && i2 < WFCF_ELEMS) {
    const int j = i2 & 15, lane = (i2 >> 4) & 31, kt = i2 >> 9;
    const int k   = kt*32 + (lane >> 4)*16 + j;
    const int col = lane & 15;                      // classes padded 10 -> 16
    wfcf[i2] = (_Float16)((col < NCLS) ? Wfc[k*NCLS + col] : 0.f);
  }
}

// -------- Persistent kernel: one WG = 16 batch rows, loops all 1000 steps --------
__global__ __launch_bounds__(256)
void plstm_persistent(const float* __restrict__ inp, const float* __restrict__ tms,
                      const float* __restrict__ Wx0, const float* __restrict__ b0,
                      const float* __restrict__ tau0, const float* __restrict__ s0,
                      const float* __restrict__ b1, const float* __restrict__ tau1,
                      const float* __restrict__ s1, const float* __restrict__ gma,
                      const float* __restrict__ bta, const float* __restrict__ bfc,
                      const _Float16* wh0f_in, const _Float16* w1f_in,
                      const _Float16* wfcf_in, float* __restrict__ out) {
  __shared__ float sZ[BT*ZW];                              // 32 KB gate pre-activations
  __shared__ alignas(64) _Float16 sStacked[KT1*32*16];     // A-frags: kt0-3 = hn, kt4-7 = h1
  __shared__ alignas(64) _Float16 sH0[KT0*32*16];          // A-frags for h0
  __shared__ float sB0[ZW], sB1[ZW], sWx0[3*ZW];
  __shared__ float sTau0[UNITS], sS0[UNITS], sTau1[UNITS], sS1[UNITS];
  __shared__ float sGam[UNITS], sBet[UNITS], sBfc[16];

  const int tid  = threadIdx.x;
  const int lane = tid & 31;
  const int wv   = tid >> 5;          // 8 waves
  const int r0   = blockIdx.x * BT;   // batch tile origin

  for (int i = tid; i < ZW; i += 256) { sB0[i] = b0[i]; sB1[i] = b1[i]; }
  for (int i = tid; i < 3*ZW; i += 256) sWx0[i] = Wx0[i];
  if (tid < UNITS) {
    sTau0[tid] = tau0[tid]; sS0[tid] = s0[tid];
    sTau1[tid] = tau1[tid]; sS1[tid] = s1[tid];
    sGam[tid]  = gma[tid];  sBet[tid] = bta[tid];
  }
  if (tid < 16) sBfc[tid] = (tid < NCLS) ? bfc[tid] : 0.f;
  for (int i = tid; i < KT1*32*16; i += 256) sStacked[i] = (_Float16)0.f;
  for (int i = tid; i < KT0*32*16; i += 256) sH0[i]      = (_Float16)0.f;

  float c0r[8], h0r[8], c1r[8], h1r[8];
  #pragma unroll
  for (int e = 0; e < 8; ++e) { c0r[e] = h0r[e] = c1r[e] = h1r[e] = 0.f; }

  // elementwise ownership: row b = tid>>4 (16 lanes/row, same half-wave), units u8*8..+7
  const int bb = tid >> 4;
  const int u8 = tid & 15;
  const int qg  = u8 & 3;
  const int ktw = u8 >> 2;
  const int gW  = qg >> 1;
  const int j0  = (qg & 1) * 8;

  // Laundered zero offset: makes weight loads loop-variant (no hoist -> no spill)
  // while keeping the base pointer a kernel argument (addrspace-inferable -> global_load).
  int lofs = 0;

  __syncthreads();

  for (int t = 0; t < TSTEPS; ++t) {
    asm volatile("" : "+s"(lofs));
    const _Float16* wh0f = wh0f_in + lofs;
    const _Float16* w1f  = w1f_in  + lofs;
    const _Float16* wfcf = wfcf_in + lofs;

    // ---- GEMM0: z = h0 @ Wh0  (A from LDS once; B batched from L2) ----
    {
      // prefetch this wave's full 16KB weight block toward near cache
      const char* pfb0 = (const char*)(wh0f + (size_t)(wv*4*KT0)*512);
      #pragma unroll
      for (int pf = 0; pf < 4; ++pf)
        __builtin_prefetch(pfb0 + pf*4096 + lane*128, 0, 3);

      v16h a0[KT0];
      #pragma unroll
      for (int kt = 0; kt < KT0; ++kt)
        a0[kt] = *(const v16h*)(sH0 + (kt*32 + lane)*16);
      #pragma unroll
      for (int q = 0; q < 4; ++q) {
        const int nt = wv*4 + q;
        const _Float16* wp = wh0f + (size_t)(nt*KT0)*512;
        v16h bf[KT0];
        #pragma unroll
        for (int kt = 0; kt < KT0; ++kt)
          bf[kt] = *(const v16h*)(wp + (kt*32 + lane)*16);
        v8f acc = {};
        #pragma unroll
        for (int kt = 0; kt < KT0; ++kt)
          acc = __builtin_amdgcn_wmma_f32_16x16x32_f16(false, a0[kt], false, bf[kt], (short)0, acc, false, false);
        const int g = lane >> 4, n = lane & 15;
        #pragma unroll
        for (int r = 0; r < 8; ++r) sZ[(g*8 + r)*ZW + nt*16 + n] = acc[r];
      }
    }
    __syncthreads();

    // ---- GATE0 + time gate + LayerNorm; emit hn (A kt0-3) and new h0 frag ----
    {
      const size_t ro = (size_t)(r0 + bb) * TSTEPS + t;
      const float x0 = inp[ro*3+0], x1 = inp[ro*3+1], x2 = inp[ro*3+2];
      const float tt = tms[ro];
      float hv[8], lsum = 0.f, lsq = 0.f;
      #pragma unroll
      for (int e = 0; e < 8; ++e) {
        const int u = u8*8 + e;
        const float zi = sZ[bb*ZW + u]       + sB0[u]       + x0*sWx0[u]       + x1*sWx0[ZW+u]       + x2*sWx0[2*ZW+u];
        const float zf = sZ[bb*ZW + 128+u]   + sB0[128+u]   + x0*sWx0[128+u]   + x1*sWx0[ZW+128+u]   + x2*sWx0[2*ZW+128+u];
        const float zg = sZ[bb*ZW + 256+u]   + sB0[256+u]   + x0*sWx0[256+u]   + x1*sWx0[ZW+256+u]   + x2*sWx0[2*ZW+256+u];
        const float zo = sZ[bb*ZW + 384+u]   + sB0[384+u]   + x0*sWx0[384+u]   + x1*sWx0[ZW+384+u]   + x2*sWx0[2*ZW+384+u];
        const float ch = sigf(zf)*c0r[e] + sigf(zi)*tanhft(zg);
        const float hh = sigf(zo)*tanhft(ch);
        const float a  = (tt - sS0[u]) / sTau0[u];
        const float phi = a - floorf(a);
        const float kk = (phi < 0.5f*R_ON) ? (phi*(2.f/R_ON))
                       : (phi < R_ON)      ? (2.f - phi*(2.f/R_ON))
                                           : (ALPHA_L*phi);
        const float hn0 = kk*hh + (1.f-kk)*h0r[e];
        c0r[e] = kk*ch + (1.f-kk)*c0r[e];
        h0r[e] = hn0; hv[e] = hn0;
        lsum += hn0; lsq += hn0*hn0;
      }
      #pragma unroll
      for (int m = 1; m < 16; m <<= 1) { lsum += __shfl_xor(lsum, m, 32); lsq += __shfl_xor(lsq, m, 32); }
      const float mu = lsum * (1.f/UNITS);
      const float rs = rsqrtf(lsq*(1.f/UNITS) - mu*mu + LN_EPS);
      union { _Float16 h[8]; uint4 v; } pkh, pkn;
      #pragma unroll
      for (int e = 0; e < 8; ++e) {
        const int u = u8*8 + e;
        pkh.h[e] = (_Float16)hv[e];
        pkn.h[e] = (_Float16)(sGam[u]*(hv[e]-mu)*rs + sBet[u]);
      }
      const int fo = (ktw*32 + gW*16 + bb)*16 + j0;
      *(uint4*)(sH0 + fo)      = pkh.v;
      *(uint4*)(sStacked + fo) = pkn.v;     // hn into kt0-3
    }
    __syncthreads();

    // ---- GEMM1: z = [hn | h1_prev] @ [Wx1; Wh1]  (K=256) ----
    {
      const char* pfb1 = (const char*)(w1f + (size_t)(wv*4*KT1)*512);
      #pragma unroll
      for (int pf = 0; pf < 8; ++pf)
        __builtin_prefetch(pfb1 + pf*4096 + lane*128, 0, 3);

      v16h a1[KT1];
      #pragma unroll
      for (int kt = 0; kt < KT1; ++kt)
        a1[kt] = *(const v16h*)(sStacked + (kt*32 + lane)*16);
      #pragma unroll
      for (int q = 0; q < 4; ++q) {
        const int nt = wv*4 + q;
        const _Float16* wp = w1f + (size_t)(nt*KT1)*512;
        v16h bf[KT1];
        #pragma unroll
        for (int kt = 0; kt < KT1; ++kt)
          bf[kt] = *(const v16h*)(wp + (kt*32 + lane)*16);
        v8f acc = {};
        #pragma unroll
        for (int kt = 0; kt < KT1; ++kt)
          acc = __builtin_amdgcn_wmma_f32_16x16x32_f16(false, a1[kt], false, bf[kt], (short)0, acc, false, false);
        const int g = lane >> 4, n = lane & 15;
        #pragma unroll
        for (int r = 0; r < 8; ++r) sZ[(g*8 + r)*ZW + nt*16 + n] = acc[r];
      }
    }
    __syncthreads();

    // ---- GATE1 + time gate; new h1 into A kt4-7 ----
    {
      const float tt = tms[(size_t)(r0 + bb) * TSTEPS + t];
      union { _Float16 h[8]; uint4 v; } pk;
      #pragma unroll
      for (int e = 0; e < 8; ++e) {
        const int u = u8*8 + e;
        const float zi = sZ[bb*ZW + u]     + sB1[u];
        const float zf = sZ[bb*ZW + 128+u] + sB1[128+u];
        const float zg = sZ[bb*ZW + 256+u] + sB1[256+u];
        const float zo = sZ[bb*ZW + 384+u] + sB1[384+u];
        const float ch = sigf(zf)*c1r[e] + sigf(zi)*tanhft(zg);
        const float hh = sigf(zo)*tanhft(ch);
        const float a  = (tt - sS1[u]) / sTau1[u];
        const float phi = a - floorf(a);
        const float kk = (phi < 0.5f*R_ON) ? (phi*(2.f/R_ON))
                       : (phi < R_ON)      ? (2.f - phi*(2.f/R_ON))
                                           : (ALPHA_L*phi);
        const float hn1 = kk*hh + (1.f-kk)*h1r[e];
        c1r[e] = kk*ch + (1.f-kk)*c1r[e];
        h1r[e] = hn1;
        pk.h[e] = (_Float16)hn1;
      }
      *(uint4*)(sStacked + ((ktw+4)*32 + gW*16 + bb)*16 + j0) = pk.v;
    }
    __syncthreads();

    // ---- FC + softmax (wave 0): probs = softmax(h1 @ Wfc + bfc) ----
    if (wv == 0) {
      v16h af[KT0], bfr[KT0];
      #pragma unroll
      for (int kt = 0; kt < KT0; ++kt) {
        af[kt]  = *(const v16h*)(sStacked + ((kt+4)*32 + lane)*16);
        bfr[kt] = *(const v16h*)(wfcf + (kt*32 + lane)*16);
      }
      v8f acc = {};
      #pragma unroll
      for (int kt = 0; kt < KT0; ++kt)
        acc = __builtin_amdgcn_wmma_f32_16x16x32_f16(false, af[kt], false, bfr[kt], (short)0, acc, false, false);
      const int g = lane >> 4, n = lane & 15;
      const float bias = sBfc[n];
      #pragma unroll
      for (int r = 0; r < 8; ++r) {
        const float v  = acc[r] + bias;
        float mx = (n < NCLS) ? v : -1e30f;
        #pragma unroll
        for (int m = 1; m < 16; m <<= 1) mx = fmaxf(mx, __shfl_xor(mx, m, 32));
        float ex = (n < NCLS) ? __expf(v - mx) : 0.f;
        float sm = ex;
        #pragma unroll
        for (int m = 1; m < 16; m <<= 1) sm += __shfl_xor(sm, m, 32);
        if (n < NCLS)
          out[((size_t)(r0 + g*8 + r) * TSTEPS + t) * NCLS + n] = ex / sm;
      }
    }
    __syncthreads();
  }
}

extern "C" void kernel_launch(void* const* d_in, const int* in_sizes, int n_in,
                              void* d_out, int out_size, void* d_ws, size_t ws_size,
                              hipStream_t stream) {
  (void)in_sizes; (void)n_in; (void)out_size; (void)ws_size;
  const float* inp  = (const float*)d_in[0];
  const float* tms  = (const float*)d_in[1];
  const float* Wx0  = (const float*)d_in[2];
  const float* Wh0  = (const float*)d_in[3];
  const float* b0   = (const float*)d_in[4];
  const float* tau0 = (const float*)d_in[5];
  const float* s0   = (const float*)d_in[6];
  const float* Wx1  = (const float*)d_in[7];
  const float* Wh1  = (const float*)d_in[8];
  const float* b1   = (const float*)d_in[9];
  const float* tau1 = (const float*)d_in[10];
  const float* s1   = (const float*)d_in[11];
  const float* gma  = (const float*)d_in[12];
  const float* bta  = (const float*)d_in[13];
  const float* Wfc  = (const float*)d_in[14];
  const float* bfc  = (const float*)d_in[15];

  _Float16* wh0f = (_Float16*)d_ws;
  _Float16* w1f  = (_Float16*)((char*)d_ws + (size_t)WH0F_ELEMS*2);
  _Float16* wfcf = (_Float16*)((char*)d_ws + (size_t)(WH0F_ELEMS + W1F_ELEMS)*2);

  const int total = WH0F_ELEMS + W1F_ELEMS + WFCF_ELEMS;
  plstm_prep<<<(total + 255)/256, 256, 0, stream>>>(Wh0, Wx1, Wh1, Wfc, wh0f, w1f, wfcf);
  plstm_persistent<<<NWG, 256, 0, stream>>>(inp, tms, Wx0, b0, tau0, s0, b1, tau1, s1,
                                            gma, bta, bfc, wh0f, w1f, wfcf, (float*)d_out);
}
